// graph_convolution_33938831573435
// MI455X (gfx1250) — matrix operand order repaired and checked by reading, not compiled, not run on hardware
//
#include <hip/hip_runtime.h>
#include <hip/hip_bf16.h>

typedef __attribute__((ext_vector_type(16))) _Float16 v16h;
typedef __attribute__((ext_vector_type(8)))  float    v8f;
typedef __attribute__((ext_vector_type(4)))  unsigned int u4x;

union HVec { u4x u[2]; v16h h; };

#define LN_EPS 1e-5f

// ---------------------------------------------------------------------------
// LayerNorm statistics over M rows x 128 cols held in LDS (row stride RS).
// Must be called by all 256 threads. M in {16, 32}.
// ---------------------------------------------------------------------------
template <int M>
__device__ __forceinline__ void ln_stats(const float* sRaw, int RS,
                                         float* sPs, float* sPq,
                                         float* sMu, float* sRsd, int t) {
  if (t < M * 8) {
    int r = t >> 3, seg = t & 7;
    const float* row = sRaw + r * RS + seg * 16;
    float s = 0.f, q = 0.f;
#pragma unroll
    for (int j = 0; j < 16; ++j) { float v = row[j]; s += v; q += v * v; }
    sPs[r * 8 + seg] = s; sPq[r * 8 + seg] = q;
  }
  __syncthreads();
  if (t < M) {
    float s = 0.f, q = 0.f;
#pragma unroll
    for (int j = 0; j < 8; ++j) { s += sPs[t * 8 + j]; q += sPq[t * 8 + j]; }
    float mu = s * (1.f / 128.f);
    float var = q * (1.f / 128.f) - mu * mu;
    sMu[t] = mu; sRsd[t] = rsqrtf(var + LN_EPS);
  }
  __syncthreads();
}

// ---------------------------------------------------------------------------
// Pack f32 weight W[K,128] (row major, z = x @ W) into WMMA B-operand layout:
// flat[((nt*KC + kc)*32 + lane)*16 + i] = W[kc*32 + (lane<16?0:8) + (i&7) + (i<8?0:16)][nt*16 + lane%16]
// ---------------------------------------------------------------------------
__global__ void pack_w_kernel(const float* __restrict__ W, _Float16* __restrict__ dst, int K) {
  int i = blockIdx.x * blockDim.x + threadIdx.x;
  int total = K * 128;
  if (i >= total) return;
  int ii = i & 15;
  int l  = (i >> 4) & 31;
  int rest = i >> 9;
  int KC = K >> 5;
  int kc = rest % KC;
  int nt = rest / KC;
  int k = kc * 32 + ((l & 16) ? 8 : 0) + (ii & 7) + ((ii & 8) ? 16 : 0);
  int n = nt * 16 + (l & 15);
  dst[i] = (_Float16)W[(size_t)k * 128 + n];
}

// f32 [rows, cols] -> f16 with destination row stride (dst already col-offset)
__global__ void cvt_kernel(const float* __restrict__ src, _Float16* __restrict__ dst,
                           int rows, int cols, int dstStride) {
  int i = blockIdx.x * blockDim.x + threadIdx.x;
  if (i >= rows * cols) return;
  int r = i / cols, c = i % cols;
  dst[(size_t)r * dstStride + c] = (_Float16)src[i];
}

// ---------------------------------------------------------------------------
// Fused per-edge 2-layer MLP + LN + ReLU + segment-max scatter.
// C = node feature width. Processes M=32 edges per barrier round; each wave
// owns one 16-col output tile and two 16-row accumulators sharing register-
// resident B operands.
// ---------------------------------------------------------------------------
template <int C>
__global__ __launch_bounds__(256)
void edge_mlp_kernel(const _Float16* __restrict__ xh, int xStrideH,
                     const int* __restrict__ srcIdx, const int* __restrict__ dstIdx,
                     const _Float16* __restrict__ wp1,
                     const float* __restrict__ b1, const float* __restrict__ g1, const float* __restrict__ be1,
                     const _Float16* __restrict__ wp2,
                     const float* __restrict__ b2, const float* __restrict__ g2, const float* __restrict__ be2,
                     int* __restrict__ aggI, int numTiles) {
  constexpr int M  = 32;       // edges per tile
  constexpr int K1 = 2 * C;
  constexpr int KC1 = K1 / 32;
  constexpr int KC2 = 4;
  constexpr int RS = 132;      // padded f32 row stride (bank-conflict avoidance)
  constexpr int UPR = K1 / 8;  // 16-byte gather units per row

  __shared__ _Float16 sA[M * K1];
  __shared__ float    sRaw[M * RS];
  __shared__ _Float16 sH[M * 128];
  __shared__ float sPs[M * 8], sPq[M * 8], sMu[M], sRsd[M];
  __shared__ int sS[M], sD[M];

  const int t = threadIdx.x;
  const int wave = t >> 5;
  const int lane = t & 31;
  const int lhalf = lane >> 4;
  const int ln = lane & 15;
  const int kb = lhalf * 8;

  // Preload this wave's weight slices (B operands) into registers.
  v16h w1[KC1], w2[KC2];
#pragma unroll
  for (int kc = 0; kc < KC1; ++kc) {
    const u4x* p = (const u4x*)(wp1 + ((size_t)((wave * KC1 + kc) * 32 + lane) << 4));
    HVec r; r.u[0] = p[0]; r.u[1] = p[1]; w1[kc] = r.h;
  }
#pragma unroll
  for (int kc = 0; kc < KC2; ++kc) {
    const u4x* p = (const u4x*)(wp2 + ((size_t)((wave * KC2 + kc) * 32 + lane) << 4));
    HVec r; r.u[0] = p[0]; r.u[1] = p[1]; w2[kc] = r.h;
  }

  for (int tile = blockIdx.x; tile < numTiles; tile += gridDim.x) {
    const int ebase = tile * M;
    if (t < M) { sS[t] = srcIdx[ebase + t]; sD[t] = dstIdx[ebase + t]; }
    __syncthreads();

    // Gather A rows = [x[dst] | x[src]] (f16), 16-byte units.
    for (int u = t; u < M * UPR; u += 256) {
      int r = u / UPR;
      int col = (u % UPR) * 8;  // half offset within row
      const _Float16* sp = (col < C) ? (xh + (size_t)sD[r] * xStrideH + col)
                                     : (xh + (size_t)sS[r] * xStrideH + (col - C));
      *(u4x*)(sA + r * K1 + col) = *(const u4x*)sp;
    }
    __syncthreads();

    // Layer 1: two 16-row accumulators share w1[kc]
    {
      v8f acc0 = {}, acc1 = {};
#pragma unroll
      for (int kc = 0; kc < KC1; ++kc) {
        const _Float16* a0 = sA + ln * K1 + kc * 32 + kb;
        const _Float16* a1 = sA + (16 + ln) * K1 + kc * 32 + kb;
        HVec r0; r0.u[0] = *(const u4x*)a0; r0.u[1] = *(const u4x*)(a0 + 16);
        HVec r1; r1.u[0] = *(const u4x*)a1; r1.u[1] = *(const u4x*)(a1 + 16);
        acc0 = __builtin_amdgcn_wmma_f32_16x16x32_f16(false, r0.h, false, w1[kc],
                                                      (short)0, acc0, false, false);
        acc1 = __builtin_amdgcn_wmma_f32_16x16x32_f16(false, r1.h, false, w1[kc],
                                                      (short)0, acc1, false, false);
      }
      const int n = wave * 16 + ln;
      const float bb = b1[n];
#pragma unroll
      for (int v = 0; v < 8; ++v) {
        sRaw[(v + lhalf * 8) * RS + n]        = acc0[v] + bb;
        sRaw[(16 + v + lhalf * 8) * RS + n]   = acc1[v] + bb;
      }
    }
    __syncthreads();

    ln_stats<M>(sRaw, RS, sPs, sPq, sMu, sRsd, t);

    // Apply LN + ReLU -> f16 hidden (two 16-row passes)
#pragma unroll
    for (int rr = 0; rr < 2; ++rr) {
      int r = (t >> 4) + rr * 16, cb = (t & 15) * 8;
      float mu = sMu[r], rs = sRsd[r];
#pragma unroll
      for (int j = 0; j < 8; ++j) {
        int c = cb + j;
        float y = (sRaw[r * RS + c] - mu) * rs * g1[c] + be1[c];
        sH[r * 128 + c] = (_Float16)fmaxf(y, 0.f);
      }
    }
    __syncthreads();

    // Layer 2
    {
      v8f acc0 = {}, acc1 = {};
#pragma unroll
      for (int kc = 0; kc < KC2; ++kc) {
        const _Float16* a0 = sH + ln * 128 + kc * 32 + kb;
        const _Float16* a1 = sH + (16 + ln) * 128 + kc * 32 + kb;
        HVec r0; r0.u[0] = *(const u4x*)a0; r0.u[1] = *(const u4x*)(a0 + 16);
        HVec r1; r1.u[0] = *(const u4x*)a1; r1.u[1] = *(const u4x*)(a1 + 16);
        acc0 = __builtin_amdgcn_wmma_f32_16x16x32_f16(false, r0.h, false, w2[kc],
                                                      (short)0, acc0, false, false);
        acc1 = __builtin_amdgcn_wmma_f32_16x16x32_f16(false, r1.h, false, w2[kc],
                                                      (short)0, acc1, false, false);
      }
      const int n = wave * 16 + ln;
      const float bb = b2[n];
#pragma unroll
      for (int v = 0; v < 8; ++v) {
        sRaw[(v + lhalf * 8) * RS + n]      = acc0[v] + bb;
        sRaw[(16 + v + lhalf * 8) * RS + n] = acc1[v] + bb;
      }
    }
    __syncthreads();

    ln_stats<M>(sRaw, RS, sPs, sPq, sMu, sRsd, t);

    // Apply LN + ReLU, scatter segment-max (non-negative => int atomicMax valid)
#pragma unroll
    for (int rr = 0; rr < 2; ++rr) {
      int r = (t >> 4) + rr * 16, cb = (t & 15) * 8;
      float mu = sMu[r], rs = sRsd[r];
      int* dstp = aggI + (size_t)sD[r] * 128;
#pragma unroll
      for (int j = 0; j < 8; ++j) {
        int c = cb + j;
        float y = (sRaw[r * RS + c] - mu) * rs * g2[c] + be2[c];
        y = fmaxf(y, 0.f);
        atomicMax(dstp + c, __float_as_int(y));
      }
    }
    __syncthreads();
  }
}

// ---------------------------------------------------------------------------
// Single ffn layer on node rows: out = [addin +] act(LN(A @ W + b)); A f16.
// Optionally also writes f16 copy of the output (for the next block's edges).
// ---------------------------------------------------------------------------
template <int KC>
__global__ __launch_bounds__(256)
void node_layer_kernel(const _Float16* __restrict__ ah, int aStrideH,
                       const _Float16* __restrict__ wp,
                       const float* __restrict__ bb, const float* __restrict__ gg, const float* __restrict__ bee,
                       const float* __restrict__ addin,
                       float* __restrict__ out32,
                       _Float16* __restrict__ out16, int o16Stride,
                       int relu, int numTiles) {
  constexpr int K = KC * 32;
  constexpr int RS = 132;
  __shared__ _Float16 sA[16 * K];
  __shared__ float sRaw[16 * RS];
  __shared__ float sPs[128], sPq[128], sMu[16], sRsd[16];

  const int t = threadIdx.x;
  const int wave = t >> 5;
  const int lane = t & 31;
  const int lhalf = lane >> 4;
  const int ln = lane & 15;
  const int kbo = lhalf * 8;

  v16h wreg[KC];
#pragma unroll
  for (int kc = 0; kc < KC; ++kc) {
    const u4x* p = (const u4x*)(wp + ((size_t)((wave * KC + kc) * 32 + lane) << 4));
    HVec r; r.u[0] = p[0]; r.u[1] = p[1]; wreg[kc] = r.h;
  }

  for (int tile = blockIdx.x; tile < numTiles; tile += gridDim.x) {
    const int nodeBase = tile * 16;
    // Stage 16 node rows of K halves
    for (int u = t; u < 2 * K; u += 256) {  // 16-byte units, K/8 per row
      int r = u / (K / 8), cu = u % (K / 8);
      *(u4x*)(sA + r * K + cu * 8) =
          *(const u4x*)(ah + (size_t)(nodeBase + r) * aStrideH + cu * 8);
    }
    __syncthreads();

    {
      v8f acc = {};
#pragma unroll
      for (int kc = 0; kc < KC; ++kc) {
        const _Float16* ab = sA + ln * K + kc * 32 + kbo;
        HVec r; r.u[0] = *(const u4x*)ab; r.u[1] = *(const u4x*)(ab + 16);
        acc = __builtin_amdgcn_wmma_f32_16x16x32_f16(false, r.h, false, wreg[kc],
                                                     (short)0, acc, false, false);
      }
      const int n = wave * 16 + ln;
      const float b0 = bb[n];
#pragma unroll
      for (int v = 0; v < 8; ++v) sRaw[(v + lhalf * 8) * RS + n] = acc[v] + b0;
    }
    __syncthreads();

    ln_stats<16>(sRaw, RS, sPs, sPq, sMu, sRsd, t);

    {
      int r = t >> 4, cb = (t & 15) * 8;
      int node = nodeBase + r;
      float mu = sMu[r], rs = sRsd[r];
#pragma unroll
      for (int j = 0; j < 8; ++j) {
        int c = cb + j;
        float y = (sRaw[r * RS + c] - mu) * rs * gg[c] + bee[c];
        if (relu) y = fmaxf(y, 0.f);
        if (addin) y += addin[(size_t)node * 128 + c];
        out32[(size_t)node * 128 + c] = y;
        if (out16) out16[(size_t)node * o16Stride + c] = (_Float16)y;
      }
    }
    __syncthreads();
  }
}

// ---------------------------------------------------------------------------
extern "C" void kernel_launch(void* const* d_in, const int* in_sizes, int n_in,
                              void* d_out, int out_size, void* d_ws, size_t ws_size,
                              hipStream_t stream) {
  (void)n_in; (void)out_size; (void)ws_size;
  const float* x0 = (const float*)d_in[0];
  const int*   ei = (const int*)d_in[1];
  const int N = in_sizes[0] / 64;     // 20000
  const int E = in_sizes[1] / 2;      // 640000
  const int* srcI = ei;
  const int* dstI = ei + E;

  // Param order: b0{msg1,msg2,upd,res}, b1{msg1,msg2,upd}; each {W,b,g,be}
  const float *W[7], *Bv[7], *Gv[7], *BE[7];
  int idx = 2;
  for (int m = 0; m < 7; ++m) {
    W[m]  = (const float*)d_in[idx++];
    Bv[m] = (const float*)d_in[idx++];
    Gv[m] = (const float*)d_in[idx++];
    BE[m] = (const float*)d_in[idx++];
  }
  const int Kdim[7] = {128, 128, 192, 64, 256, 128, 256};

  char* wsb = (char*)d_ws;
  size_t cur = 0;
  auto take = [&](size_t bytes) -> void* {
    cur = (cur + 255) & ~(size_t)255;
    void* p = wsb + cur;
    cur += bytes;
    return p;
  };
  _Float16* wp[7];
  for (int m = 0; m < 7; ++m) wp[m] = (_Float16*)take((size_t)Kdim[m] * 128 * 2);
  _Float16* xa  = (_Float16*)take((size_t)N * 192 * 2);  // [x0_h(64) | agg_h(128)]
  _Float16* xb  = (_Float16*)take((size_t)N * 256 * 2);  // [x1_h(128)| agg_h(128)]
  float*    agg = (float*)take((size_t)N * 128 * 4);
  float*    idn = (float*)take((size_t)N * 128 * 4);
  float*    x1  = (float*)take((size_t)N * 128 * 4);
  float*    out = (float*)d_out;

  for (int m = 0; m < 7; ++m) {
    int tot = Kdim[m] * 128;
    pack_w_kernel<<<(tot + 255) / 256, 256, 0, stream>>>(W[m], wp[m], Kdim[m]);
  }
  // x0 -> xa[:, :64] (f16)
  cvt_kernel<<<(N * 64 + 255) / 256, 256, 0, stream>>>(x0, xa, N, 64, 192);

  // ---- Block 0 ----
  hipMemsetAsync(agg, 0, (size_t)N * 128 * 4, stream);
  edge_mlp_kernel<64><<<4096, 256, 0, stream>>>(
      xa, 192, srcI, dstI,
      wp[0], Bv[0], Gv[0], BE[0],
      wp[1], Bv[1], Gv[1], BE[1],
      (int*)agg, E / 32);
  // identity = LN(x0 @ Wres + b)
  node_layer_kernel<2><<<1250, 256, 0, stream>>>(
      xa, 192, wp[3], Bv[3], Gv[3], BE[3],
      nullptr, idn, nullptr, 0, /*relu=*/0, N / 16);
  // agg -> xa[:, 64:192]
  cvt_kernel<<<(N * 128 + 255) / 256, 256, 0, stream>>>(agg, xa + 64, N, 128, 192);
  // x1 = identity + ffn(concat(x0, agg) @ Wupd); also write f16 into xb[:, :128]
  node_layer_kernel<6><<<1250, 256, 0, stream>>>(
      xa, 192, wp[2], Bv[2], Gv[2], BE[2],
      idn, x1, xb, 256, /*relu=*/1, N / 16);

  // ---- Block 1 ----
  hipMemsetAsync(agg, 0, (size_t)N * 128 * 4, stream);
  edge_mlp_kernel<128><<<4096, 256, 0, stream>>>(
      xb, 256, srcI, dstI,
      wp[4], Bv[4], Gv[4], BE[4],
      wp[5], Bv[5], Gv[5], BE[5],
      (int*)agg, E / 32);
  cvt_kernel<<<(N * 128 + 255) / 256, 256, 0, stream>>>(agg, xb + 128, N, 128, 256);
  // out = x1 + ffn(concat(x1, agg) @ Wupd)
  node_layer_kernel<8><<<1250, 256, 0, stream>>>(
      xb, 256, wp[6], Bv[6], Gv[6], BE[6],
      x1, out, nullptr, 0, /*relu=*/1, N / 16);
}
